// CMRHead_13881334301280
// MI455X (gfx1250) — matrix operand order repaired and checked by reading, not compiled, not run on hardware
//
#include <hip/hip_runtime.h>
#include <hip/hip_bf16.h>
#include <stdint.h>

typedef __attribute__((ext_vector_type(16))) __bf16 bf16x16;
typedef __attribute__((ext_vector_type(8)))  __bf16 bf16x8;
typedef __attribute__((ext_vector_type(8)))  float  f32x8;
typedef __attribute__((ext_vector_type(4)))  int    i32x4;

#define BATCH 4096
#define INF   10338   // 1723*6
#define KPAD  10368   // multiple of 64
#define HDIM  1024
#define NOUT  229     // 216 + 10 + 3
#define NPAD  256
#define EPSV  1e-5f

// GEMM tiling: 128x64 C tile per 256-thread block, K staged 64 at a time,
// double-buffered LDS. 8 wave32s, each owns a 32x32 quadrant (4 accumulators).
#define BM 128
#define BN 64
#define BK 64

#if defined(__AMDGCN__) && \
    __has_builtin(__builtin_amdgcn_global_load_async_to_lds_b128) && \
    __has_builtin(__builtin_amdgcn_s_wait_asynccnt)
#define HAVE_ASYNC_LDS 1
#else
#define HAVE_ASYNC_LDS 0
#endif

__device__ __forceinline__ __bf16 f2bf(float f) {
  union { float f; uint32_t u; } v; v.f = f;
  uint32_t r = v.u + 0x7FFFu + ((v.u >> 16) & 1u);  // round-to-nearest-even
  union { uint16_t s; __bf16 b; } o; o.s = (uint16_t)(r >> 16);
  return o.b;
}

#if HAVE_ASYNC_LDS
// Builtin signature (per clang-22 diagnostic): (v4i AS1*, v4i AS3*, imm, imm)
__device__ __forceinline__ void async_cp16(const __bf16* g, __bf16* l) {
  __builtin_amdgcn_global_load_async_to_lds_b128(
      (__attribute__((address_space(1))) i32x4*)(i32x4*)(void*)const_cast<__bf16*>(g),
      (__attribute__((address_space(3))) i32x4*)(i32x4*)(void*)l,
      0, 0);
}
#endif

// Assemble ISA 7.12.2 fragments from LDS (all contiguous 16B reads).
// A (16-bit 16x32): lane ln=row M, kg=lane>>4: K = kg*8 + {0..7, 16..23}.
__device__ __forceinline__ bf16x16 frag_a(const __bf16* p) {
  bf16x8 lo = *(const bf16x8*)p;
  bf16x8 hi = *(const bf16x8*)(p + 16);
  bf16x16 r;
#pragma unroll
  for (int e = 0; e < 8; ++e) { r[e] = lo[e]; r[e + 8] = hi[e]; }
  return r;
}
// B (32x16): lane ln=col N, K = kg*8 + {0..7, 16..23} (W^T layout).
__device__ __forceinline__ bf16x16 frag_b(const __bf16* p) {
  bf16x8 lo = *(const bf16x8*)p;
  bf16x8 hi = *(const bf16x8*)(p + 16);
  bf16x16 r;
#pragma unroll
  for (int e = 0; e < 8; ++e) { r[e] = lo[e]; r[e + 8] = hi[e]; }
  return r;
}

// ---------------------------------------------------------------------------
// fp32 -> bf16 with zero padding (row-major, for the activation matrix x).
// ---------------------------------------------------------------------------
__global__ void cvt_pad(const float* __restrict__ src, __bf16* __restrict__ dst,
                        int sm, int sn, int dm, int dn) {
  size_t tot = (size_t)dm * dn;
  for (size_t i = (size_t)blockIdx.x * blockDim.x + threadIdx.x; i < tot;
       i += (size_t)gridDim.x * blockDim.x) {
    int r = (int)(i / dn), c = (int)(i % dn);
    float v = (r < sm && c < sn) ? src[(size_t)r * sn + c] : 0.0f;
    dst[i] = f2bf(v);
  }
}

// fp32 [K][N] -> bf16 transposed [N][Kpad] (zero padded). One-time cost;
// makes all GEMM B staging / fragment loads contiguous (no LDS transpose).
__global__ void cvt_pad_T(const float* __restrict__ src, __bf16* __restrict__ dst,
                          int sk, int sn, int dk, int dn) {
  size_t tot = (size_t)dn * dk;
  for (size_t i = (size_t)blockIdx.x * blockDim.x + threadIdx.x; i < tot;
       i += (size_t)gridDim.x * blockDim.x) {
    int n = (int)(i / dk), k = (int)(i % dk);
    float v = (k < sk && n < sn) ? src[(size_t)k * sn + n] : 0.0f;
    dst[i] = f2bf(v);
  }
}

__global__ void pad_bias(const float* __restrict__ src, float* __restrict__ dst,
                         int n, int np) {
  int i = blockIdx.x * blockDim.x + threadIdx.x;
  if (i < np) dst[i] = (i < n) ? src[i] : 0.0f;
}

// ---------------------------------------------------------------------------
// C(f32, MxN) = A(bf16, MxK) @ Bt(bf16, NxK)^T + bias
// ---------------------------------------------------------------------------
__global__ __launch_bounds__(256) void wmma_gemm_bf16(
    const __bf16* __restrict__ A, const __bf16* __restrict__ Bt,
    const float* __restrict__ bias, float* __restrict__ C,
    int M, int N, int K) {
  __shared__ __bf16 As[2][BM * BK];  // [m][k]   2 x 16 KB
  __shared__ __bf16 Bs[2][BN * BK];  // [n][k]   2 x  8 KB

  const int tid  = threadIdx.x;
  const int row0 = blockIdx.y * BM;
  const int col0 = blockIdx.x * BN;
  const int wave = tid >> 5;
  const int lane = tid & 31;
  const int wm = wave & 3;    // row quadrant: 32 rows
  const int wn = wave >> 2;   // col quadrant: 32 cols
  const int kg = lane >> 4;
  const int ln = lane & 15;

  // staging assignment: A 128x64 -> 32 k per thread; Bt 64x64 -> 16 k per thread
  const int ar = tid >> 1, ak = (tid & 1) << 5;
  const int br = tid >> 2, bk = (tid & 3) << 4;
  const __bf16* Ag = A  + (size_t)(row0 + ar) * K + ak;
  const __bf16* Bg = Bt + (size_t)(col0 + br) * K + bk;

  f32x8 acc00 = {0,0,0,0,0,0,0,0};
  f32x8 acc01 = acc00, acc10 = acc00, acc11 = acc00;

  const int nk = K >> 6;

  // ---- prologue: stage tile 0 into buffer 0 ----
#if HAVE_ASYNC_LDS
  {
    __bf16* la = &As[0][ar * BK + ak];
#pragma unroll
    for (int j = 0; j < 4; ++j) async_cp16(Ag + j * 8, la + j * 8);
    __bf16* lb = &Bs[0][br * BK + bk];
#pragma unroll
    for (int j = 0; j < 2; ++j) async_cp16(Bg + j * 8, lb + j * 8);
    __builtin_amdgcn_s_wait_asynccnt(0);
  }
#else
  {
    __bf16* la = &As[0][ar * BK + ak];
#pragma unroll
    for (int j = 0; j < 4; ++j) *(bf16x8*)(la + j * 8) = *(const bf16x8*)(Ag + j * 8);
    __bf16* lb = &Bs[0][br * BK + bk];
#pragma unroll
    for (int j = 0; j < 2; ++j) *(bf16x8*)(lb + j * 8) = *(const bf16x8*)(Bg + j * 8);
  }
#endif
  __syncthreads();

  for (int kt = 0; kt < nk; ++kt) {
    const int cur = kt & 1;
    const int nxt = cur ^ 1;

#if HAVE_ASYNC_LDS
    // issue async global->LDS copies for tile kt+1 (overlaps with WMMAs below)
    if (kt + 1 < nk) {
      const __bf16* ga = Ag + (size_t)(kt + 1) * BK;
      __bf16* la = &As[nxt][ar * BK + ak];
#pragma unroll
      for (int j = 0; j < 4; ++j) async_cp16(ga + j * 8, la + j * 8);
      const __bf16* gb = Bg + (size_t)(kt + 1) * BK;
      __bf16* lb = &Bs[nxt][br * BK + bk];
#pragma unroll
      for (int j = 0; j < 2; ++j) async_cp16(gb + j * 8, lb + j * 8);
    }
#else
    bf16x8 avr[4], bvr[2];
    if (kt + 1 < nk) {
      const __bf16* ga = Ag + (size_t)(kt + 1) * BK;
#pragma unroll
      for (int j = 0; j < 4; ++j) avr[j] = *(const bf16x8*)(ga + j * 8);
      const __bf16* gb = Bg + (size_t)(kt + 1) * BK;
#pragma unroll
      for (int j = 0; j < 2; ++j) bvr[j] = *(const bf16x8*)(gb + j * 8);
    }
#endif

    // ---- compute: two K=32 WMMA steps over the 64-deep tile ----
#pragma unroll
    for (int ks = 0; ks < 2; ++ks) {
      const __bf16* ap0 = &As[cur][(wm * 32 + ln) * BK + ks * 32 + kg * 8];
      const __bf16* bp0 = &Bs[cur][(wn * 32 + ln) * BK + ks * 32 + kg * 8];
      bf16x16 a0 = frag_a(ap0);
      bf16x16 a1 = frag_a(ap0 + 16 * BK);
      bf16x16 b0 = frag_b(bp0);
      bf16x16 b1 = frag_b(bp0 + 16 * BK);
      acc00 = __builtin_amdgcn_wmma_f32_16x16x32_bf16(false, a0, false, b0,
                                                      (short)0, acc00, false, false);
      acc01 = __builtin_amdgcn_wmma_f32_16x16x32_bf16(false, a0, false, b1,
                                                      (short)0, acc01, false, false);
      acc10 = __builtin_amdgcn_wmma_f32_16x16x32_bf16(false, a1, false, b0,
                                                      (short)0, acc10, false, false);
      acc11 = __builtin_amdgcn_wmma_f32_16x16x32_bf16(false, a1, false, b1,
                                                      (short)0, acc11, false, false);
    }

#if HAVE_ASYNC_LDS
    __builtin_amdgcn_s_wait_asynccnt(0);  // tile kt+1 fully landed in LDS
#else
    if (kt + 1 < nk) {
      __bf16* la = &As[nxt][ar * BK + ak];
#pragma unroll
      for (int j = 0; j < 4; ++j) *(bf16x8*)(la + j * 8) = avr[j];
      __bf16* lb = &Bs[nxt][br * BK + bk];
#pragma unroll
      for (int j = 0; j < 2; ++j) *(bf16x8*)(lb + j * 8) = bvr[j];
    }
#endif
    __syncthreads();
  }

  // C/D layout: VGPR r -> M = r + 8*kg ; N = ln.
  const int r0 = row0 + wm * 32 + kg * 8;
  const int c0 = col0 + wn * 32 + ln;
  const float bi0 = bias[c0], bi1 = bias[c0 + 16];
#pragma unroll
  for (int r = 0; r < 8; ++r) {
    C[(size_t)(r0 + r) * N + c0]           = acc00[r] + bi0;
    C[(size_t)(r0 + r) * N + c0 + 16]      = acc01[r] + bi1;
    C[(size_t)(r0 + 16 + r) * N + c0]      = acc10[r] + bi0;
    C[(size_t)(r0 + 16 + r) * N + c0 + 16] = acc11[r] + bi1;
  }
}

// ---------------------------------------------------------------------------
// BatchNorm column stats over M rows (population variance, matches jnp.var).
// ---------------------------------------------------------------------------
__global__ __launch_bounds__(256) void bn_stats(const float* __restrict__ a,
    float* __restrict__ mean, float* __restrict__ rstd, int M, int N) {
  __shared__ float ss[256], sq[256];
  int col = blockIdx.x;
  float s = 0.f, q = 0.f;
  for (int r = threadIdx.x; r < M; r += 256) {
    float v = a[(size_t)r * N + col];
    s += v; q += v * v;
  }
  ss[threadIdx.x] = s; sq[threadIdx.x] = q;
  __syncthreads();
  for (int o = 128; o > 0; o >>= 1) {
    if (threadIdx.x < o) {
      ss[threadIdx.x] += ss[threadIdx.x + o];
      sq[threadIdx.x] += sq[threadIdx.x + o];
    }
    __syncthreads();
  }
  if (threadIdx.x == 0) {
    float mu  = ss[0] / (float)M;
    float var = sq[0] / (float)M - mu * mu;
    mean[col] = mu;
    rstd[col] = rsqrtf(var + EPSV);
  }
}

// ---------------------------------------------------------------------------
// Fused BN apply (+optional residual) + ReLU; writes fp32 (opt) and bf16.
// ---------------------------------------------------------------------------
__global__ void bn_apply(const float* __restrict__ a, const float* __restrict__ mean,
    const float* __restrict__ rstd, const float* __restrict__ g,
    const float* __restrict__ be, const float* __restrict__ resid,
    float* __restrict__ outf, __bf16* __restrict__ outb, int M, int N) {
  size_t tot = (size_t)M * N;
  for (size_t i = (size_t)blockIdx.x * blockDim.x + threadIdx.x; i < tot;
       i += (size_t)gridDim.x * blockDim.x) {
    int c = (int)(i % N);
    float v = g[c] * (a[i] - mean[c]) * rstd[c] + be[c];
    if (resid) v += resid[i];
    v = fmaxf(v, 0.0f);
    if (outf) outf[i] = v;
    outb[i] = f2bf(v);
  }
}

// ---------------------------------------------------------------------------
// 3x3 polar decomposition (== U@Vh from SVD) via scaled Newton iteration,
// then multiply by det(R) like the reference. One thread per matrix.
// ---------------------------------------------------------------------------
__device__ __forceinline__ float det3(const float* m) {
  return m[0] * (m[4] * m[8] - m[5] * m[7])
       - m[1] * (m[3] * m[8] - m[5] * m[6])
       + m[2] * (m[3] * m[7] - m[4] * m[6]);
}

__global__ void polar_svd(const float* __restrict__ o4, float* __restrict__ Rout) {
  int idx = blockIdx.x * blockDim.x + threadIdx.x;
  if (idx >= BATCH * 24) return;
  int b = idx / 24, j = idx % 24;
  const float* src = o4 + (size_t)b * NPAD + j * 9;
  float X[9];
#pragma unroll
  for (int k = 0; k < 9; ++k) X[k] = src[k];

#pragma unroll 1
  for (int it = 0; it < 12; ++it) {
    float d  = det3(X);
    float ad = fabsf(d);
    if (ad < 1e-25f) {  // regularize near-singular
      X[0] += 1e-6f; X[4] += 1e-6f; X[8] += 1e-6f;
      d = det3(X); ad = fabsf(d);
    }
    float c0 = X[4]*X[8] - X[5]*X[7];
    float c1 = X[5]*X[6] - X[3]*X[8];
    float c2 = X[3]*X[7] - X[4]*X[6];
    float c3 = X[2]*X[7] - X[1]*X[8];
    float c4 = X[0]*X[8] - X[2]*X[6];
    float c5 = X[1]*X[6] - X[0]*X[7];
    float c6 = X[1]*X[5] - X[2]*X[4];
    float c7 = X[2]*X[3] - X[0]*X[5];
    float c8 = X[0]*X[4] - X[1]*X[3];
    float inv_d = 1.0f / d;
    float gsc = 1.0f / cbrtf(ad);   // determinant scaling accelerates convergence
    float igs = 1.0f / gsc;
    float Y[9] = {c0, c1, c2, c3, c4, c5, c6, c7, c8};
#pragma unroll
    for (int k = 0; k < 9; ++k) X[k] = 0.5f * (gsc * X[k] + igs * inv_d * Y[k]);
  }
  float d = det3(X);
#pragma unroll
  for (int k = 0; k < 9; ++k)
    Rout[(size_t)b * 216 + j * 9 + k] = X[k] * d;
}

// betas (cols 216..225) and camera (226..228) into the concatenated output.
__global__ void tail_copy(const float* __restrict__ o4, float* __restrict__ out) {
  int i = blockIdx.x * blockDim.x + threadIdx.x;
  if (i >= BATCH * 13) return;
  int b = i / 13, c = i % 13;
  float v = o4[(size_t)b * NPAD + 216 + c];
  size_t base = (size_t)BATCH * 216;
  if (c < 10) out[base + (size_t)b * 10 + c] = v;
  else        out[base + (size_t)BATCH * 10 + (size_t)b * 3 + (c - 10)] = v;
}

// ---------------------------------------------------------------------------
extern "C" void kernel_launch(void* const* d_in, const int* in_sizes, int n_in,
                              void* d_out, int out_size, void* d_ws, size_t ws_size,
                              hipStream_t stream) {
  (void)in_sizes; (void)n_in; (void)out_size; (void)ws_size;
  const float* x    = (const float*)d_in[0];
  const float* w1   = (const float*)d_in[1];
  const float* b1   = (const float*)d_in[2];
  const float* g1   = (const float*)d_in[3];
  const float* be1  = (const float*)d_in[4];
  const float* w2a  = (const float*)d_in[5];
  const float* b2a  = (const float*)d_in[6];
  const float* g2a  = (const float*)d_in[7];
  const float* be2a = (const float*)d_in[8];
  const float* w2b  = (const float*)d_in[9];
  const float* b2b  = (const float*)d_in[10];
  const float* g2b  = (const float*)d_in[11];
  const float* be2b = (const float*)d_in[12];
  const float* w3a  = (const float*)d_in[13];
  const float* b3a  = (const float*)d_in[14];
  const float* g3a  = (const float*)d_in[15];
  const float* be3a = (const float*)d_in[16];
  const float* w3b  = (const float*)d_in[17];
  const float* b3b  = (const float*)d_in[18];
  const float* g3b  = (const float*)d_in[19];
  const float* be3b = (const float*)d_in[20];
  const float* w4   = (const float*)d_in[21];
  const float* b4   = (const float*)d_in[22];

  char* ws = (char*)d_ws;
  size_t off = 0;
  auto alloc = [&](size_t bytes) -> void* {
    off = (off + 255) & ~(size_t)255;
    void* p = ws + off;
    off += bytes;
    return p;
  };
  __bf16* xb   = (__bf16*)alloc((size_t)BATCH * KPAD * 2);
  __bf16* w1t  = (__bf16*)alloc((size_t)HDIM * KPAD * 2);   // [N][K] transposed
  __bf16* w2at = (__bf16*)alloc((size_t)HDIM * HDIM * 2);
  __bf16* w2bt = (__bf16*)alloc((size_t)HDIM * HDIM * 2);
  __bf16* w3at = (__bf16*)alloc((size_t)HDIM * HDIM * 2);
  __bf16* w3bt = (__bf16*)alloc((size_t)HDIM * HDIM * 2);
  __bf16* w4t  = (__bf16*)alloc((size_t)NPAD * HDIM * 2);
  float*  b4p  = (float*)alloc((size_t)NPAD * 4);
  float*  pre  = (float*)alloc((size_t)BATCH * HDIM * 4);
  float*  hf   = (float*)alloc((size_t)BATCH * HDIM * 4);
  __bf16* hb   = (__bf16*)alloc((size_t)BATCH * HDIM * 2);
  __bf16* tb   = (__bf16*)alloc((size_t)BATCH * HDIM * 2);
  float*  o4   = (float*)alloc((size_t)BATCH * NPAD * 4);
  float*  mu   = (float*)alloc((size_t)HDIM * 4);
  float*  rs   = (float*)alloc((size_t)HDIM * 4);

  // --- precision conversion; weights transposed to [N][Kpad] bf16 ---
  cvt_pad  <<<4096, 256, 0, stream>>>(x,   xb,   BATCH, INF,  BATCH, KPAD);
  cvt_pad_T<<<2048, 256, 0, stream>>>(w1,  w1t,  INF,  HDIM,  KPAD,  HDIM);
  cvt_pad_T<<<1024, 256, 0, stream>>>(w2a, w2at, HDIM, HDIM,  HDIM,  HDIM);
  cvt_pad_T<<<1024, 256, 0, stream>>>(w2b, w2bt, HDIM, HDIM,  HDIM,  HDIM);
  cvt_pad_T<<<1024, 256, 0, stream>>>(w3a, w3at, HDIM, HDIM,  HDIM,  HDIM);
  cvt_pad_T<<<1024, 256, 0, stream>>>(w3b, w3bt, HDIM, HDIM,  HDIM,  HDIM);
  cvt_pad_T<<<256,  256, 0, stream>>>(w4,  w4t,  HDIM, NOUT,  HDIM,  NPAD);
  pad_bias<<<1, 256, 0, stream>>>(b4, b4p, NOUT, NPAD);

  dim3 gH(HDIM / BN, BATCH / BM);   // (16, 32)
  dim3 gO(NPAD / BN, BATCH / BM);   // (4, 32)

  // --- layer 1: relu(bn(x @ w1 + b1)) ---
  wmma_gemm_bf16<<<gH, 256, 0, stream>>>(xb, w1t, b1, pre, BATCH, HDIM, KPAD);
  bn_stats<<<HDIM, 256, 0, stream>>>(pre, mu, rs, BATCH, HDIM);
  bn_apply<<<2048, 256, 0, stream>>>(pre, mu, rs, g1, be1, nullptr, hf, hb, BATCH, HDIM);

  // --- resblock 2 ---
  wmma_gemm_bf16<<<gH, 256, 0, stream>>>(hb, w2at, b2a, pre, BATCH, HDIM, HDIM);
  bn_stats<<<HDIM, 256, 0, stream>>>(pre, mu, rs, BATCH, HDIM);
  bn_apply<<<2048, 256, 0, stream>>>(pre, mu, rs, g2a, be2a, nullptr, nullptr, tb, BATCH, HDIM);
  wmma_gemm_bf16<<<gH, 256, 0, stream>>>(tb, w2bt, b2b, pre, BATCH, HDIM, HDIM);
  bn_stats<<<HDIM, 256, 0, stream>>>(pre, mu, rs, BATCH, HDIM);
  bn_apply<<<2048, 256, 0, stream>>>(pre, mu, rs, g2b, be2b, hf, hf, hb, BATCH, HDIM);

  // --- resblock 3 ---
  wmma_gemm_bf16<<<gH, 256, 0, stream>>>(hb, w3at, b3a, pre, BATCH, HDIM, HDIM);
  bn_stats<<<HDIM, 256, 0, stream>>>(pre, mu, rs, BATCH, HDIM);
  bn_apply<<<2048, 256, 0, stream>>>(pre, mu, rs, g3a, be3a, nullptr, nullptr, tb, BATCH, HDIM);
  wmma_gemm_bf16<<<gH, 256, 0, stream>>>(tb, w3bt, b3b, pre, BATCH, HDIM, HDIM);
  bn_stats<<<HDIM, 256, 0, stream>>>(pre, mu, rs, BATCH, HDIM);
  bn_apply<<<2048, 256, 0, stream>>>(pre, mu, rs, g3b, be3b, hf, hf, hb, BATCH, HDIM);

  // --- output layer: h @ w4 + b4 (N padded 229 -> 256) ---
  wmma_gemm_bf16<<<gO, 256, 0, stream>>>(hb, w4t, b4p, o4, BATCH, NPAD, HDIM);

  // --- SVD-orthogonalize rotations; copy betas/camera ---
  polar_svd<<<(BATCH * 24 + 255) / 256, 256, 0, stream>>>(o4, (float*)d_out);
  tail_copy<<<(BATCH * 13 + 255) / 256, 256, 0, stream>>>(o4, (float*)d_out);
}
